// GNN_89824946028589
// MI455X (gfx1250) — matrix operand order repaired and checked by reading, not compiled, not run on hardware
//
#include <hip/hip_runtime.h>
#include <stdint.h>

#define N_NODES 50000
#define N_PAD   50048          // 391 * 128
#define N_EDGES 200000
#define DIM     702
#define DIM2    1404
#define NGRAPH  512
#define NTASKS  128
#define LD7     704            // K-pad of 702 (multiple of 32)
#define LD14    1408           // K-pad of 1404
#define BN_EPS  1e-5f

// ---------------- WMMA types ----------------
typedef __attribute__((ext_vector_type(16))) __bf16 v16bf;
typedef __attribute__((ext_vector_type(8)))  float  v8f;

union Frag {
    uint32_t u[8];
    v16bf    v;
};

__device__ __forceinline__ unsigned short f2bf(float f) {
    unsigned int u = __float_as_uint(f);
    u += 0x7FFFu + ((u >> 16) & 1u);   // round-to-nearest-even
    return (unsigned short)(u >> 16);
}

// ---------------- GEMM: C[M x Ncols] = A[M x K] (bf16) @ Wt[Ncols x K]^T (bf16) + bias ----------
// block tile 128x128, BK=32, 256 threads = 8 wave32s, wave tile 32x64 (2x4 wmma accumulators).
// Double-buffered CDNA5 async global->LDS DMA (global_load_async_to_lds_b128) pipelined
// against the ds_load_b128 + v_wmma compute phase; s_wait_asynccnt drains exactly one tile.
#define BM 128
#define BN 128
#define BK 32
#define LSTR 40   // LDS row stride in bf16 units (padded: 80B, multiple of 16B)

#define FLAG_RELU 1
#define FLAG_BF16 2

__device__ __forceinline__ void compute_tile(
    const unsigned short* __restrict__ asp, const unsigned short* __restrict__ bsp,
    int wm, int wn, int frow, int half, v8f acc[2][4])
{
    Frag fa[2];
    #pragma unroll
    for (int i = 0; i < 2; ++i) {
        const int r = wm * 32 + i * 16 + frow;
        // A layout (16-bit 16x32): lanes0-15 K=0..7,16..23 ; lanes16-31 K=8..15,24..31
        const uint32_t* p0 = reinterpret_cast<const uint32_t*>(&asp[r * LSTR + 8 * half]);
        const uint32_t* p1 = reinterpret_cast<const uint32_t*>(&asp[r * LSTR + 16 + 8 * half]);
        fa[i].u[0] = p0[0]; fa[i].u[1] = p0[1]; fa[i].u[2] = p0[2]; fa[i].u[3] = p0[3];
        fa[i].u[4] = p1[0]; fa[i].u[5] = p1[1]; fa[i].u[6] = p1[2]; fa[i].u[7] = p1[3];
    }
    Frag fb[4];
    #pragma unroll
    for (int j = 0; j < 4; ++j) {
        const int c = wn * 64 + j * 16 + frow;
        // B layout (16-bit 32x16): lanes0-15 K=0..7,16..23 ; lanes16-31 K=8..15,24..31
        const uint32_t* p = reinterpret_cast<const uint32_t*>(&bsp[c * LSTR + 8 * half]);
        const uint32_t* q = reinterpret_cast<const uint32_t*>(&bsp[c * LSTR + 16 + 8 * half]);
        #pragma unroll
        for (int e = 0; e < 4; ++e) { fb[j].u[e] = p[e]; fb[j].u[4 + e] = q[e]; }
    }
    #pragma unroll
    for (int i = 0; i < 2; ++i)
        #pragma unroll
        for (int j = 0; j < 4; ++j)
            acc[i][j] = __builtin_amdgcn_wmma_f32_16x16x32_bf16(
                false, fa[i].v, false, fb[j].v, (short)0, acc[i][j], false, false);
}

__global__ __launch_bounds__(256) void gemm_bf16_kernel(
    const unsigned short* __restrict__ A, int ldA,     // [rowPad x ldA] bf16
    const unsigned short* __restrict__ Wt, int ldW,    // [colPad x ldW] bf16 (row = out col)
    void* __restrict__ Cv, int ldC, int colOff, int Ncols,
    const float* __restrict__ bias, int flags, int kTiles)
{
    __shared__ unsigned short As[2 * BM * LSTR];
    __shared__ unsigned short Bs[2 * BN * LSTR];

    const int tid  = threadIdx.x;
    const int lane = tid & 31;
    const int wid  = tid >> 5;      // wave32
    const int wm   = wid & 3;       // rows wm*32
    const int wn   = wid >> 2;      // cols wn*64
    const int rowBase = blockIdx.x * BM;
    const int colBase = blockIdx.y * BN;

    // ---- async tile-fill mapping: 512 chunks of 16B per tile, 2 per thread ----
    // chunk c: tile row = c>>2, k-offset = (c&3)*8 bf16
    const int c0 = tid;
    const int c1 = tid + 256;
    const int kOffE = (c0 & 3) * 8;                 // same low bits for c0 and c1

    const uint32_t asBase = (uint32_t)(uintptr_t)(&As[0]);
    const uint32_t bsBase = (uint32_t)(uintptr_t)(&Bs[0]);
    const uint32_t off0 = (uint32_t)(((c0 >> 2) * LSTR + kOffE) * 2);
    const uint32_t off1 = (uint32_t)(((c1 >> 2) * LSTR + kOffE) * 2);
    const uint32_t BUFB = (uint32_t)(BM * LSTR * 2);    // bytes per tile buffer

    const uint32_t lA0_b0 = asBase + off0,        lA1_b0 = asBase + off1;
    const uint32_t lB0_b0 = bsBase + off0,        lB1_b0 = bsBase + off1;
    const uint32_t lA0_b1 = lA0_b0 + BUFB,        lA1_b1 = lA1_b0 + BUFB;
    const uint32_t lB0_b1 = lB0_b0 + BUFB,        lB1_b1 = lB1_b0 + BUFB;

    const unsigned short* gA0 = A  + (size_t)(rowBase + (c0 >> 2)) * ldA + kOffE;
    const unsigned short* gA1 = A  + (size_t)(rowBase + (c1 >> 2)) * ldA + kOffE;
    const unsigned short* gB0 = Wt + (size_t)(colBase + (c0 >> 2)) * ldW + kOffE;
    const unsigned short* gB1 = Wt + (size_t)(colBase + (c1 >> 2)) * ldW + kOffE;

    const unsigned short* asp0 = &As[0];
    const unsigned short* asp1 = &As[BM * LSTR];
    const unsigned short* bsp0 = &Bs[0];
    const unsigned short* bsp1 = &Bs[BN * LSTR];

#define ISSUE_TILE(LA0, LA1, LB0, LB1)                                         \
    do {                                                                       \
        asm volatile(                                                          \
            "global_load_async_to_lds_b128 %0, %4, off\n\t"                    \
            "global_load_async_to_lds_b128 %1, %5, off\n\t"                    \
            "global_load_async_to_lds_b128 %2, %6, off\n\t"                    \
            "global_load_async_to_lds_b128 %3, %7, off"                        \
            :                                                                  \
            : "v"(LA0), "v"(LA1), "v"(LB0), "v"(LB1),                          \
              "v"(gA0), "v"(gA1), "v"(gB0), "v"(gB1)                           \
            : "memory");                                                       \
        gA0 += BK; gA1 += BK; gB0 += BK; gB1 += BK;                            \
    } while (0)

    v8f acc[2][4];
    #pragma unroll
    for (int i = 0; i < 2; ++i)
        #pragma unroll
        for (int j = 0; j < 4; ++j)
            #pragma unroll
            for (int v = 0; v < 8; ++v) acc[i][j][v] = 0.0f;

    const int frow = lane & 15;
    const int half = lane >> 4;

    // ---- software pipeline: tile 0 in flight before the loop (kTiles is even, >= 4) ----
    ISSUE_TILE(lA0_b0, lA1_b0, lB0_b0, lB1_b0);

    for (int kt = 0; kt < kTiles - 2; kt += 2) {
        ISSUE_TILE(lA0_b1, lA1_b1, lB0_b1, lB1_b1);          // tile kt+1 -> buf1
        asm volatile("s_wait_asynccnt 0x4" ::: "memory");     // tile kt landed
        __syncthreads();
        compute_tile(asp0, bsp0, wm, wn, frow, half, acc);
        __syncthreads();

        ISSUE_TILE(lA0_b0, lA1_b0, lB0_b0, lB1_b0);          // tile kt+2 -> buf0
        asm volatile("s_wait_asynccnt 0x4" ::: "memory");     // tile kt+1 landed
        __syncthreads();
        compute_tile(asp1, bsp1, wm, wn, frow, half, acc);
        __syncthreads();
    }
    // ---- tail: tile kTiles-2 in flight in buf0 ----
    ISSUE_TILE(lA0_b1, lA1_b1, lB0_b1, lB1_b1);              // last tile -> buf1
    asm volatile("s_wait_asynccnt 0x4" ::: "memory");
    __syncthreads();
    compute_tile(asp0, bsp0, wm, wn, frow, half, acc);
    __syncthreads();
    asm volatile("s_wait_asynccnt 0x0" ::: "memory");
    __syncthreads();
    compute_tile(asp1, bsp1, wm, wn, frow, half, acc);
#undef ISSUE_TILE

    // ---- epilogue: bias + optional ReLU, f32 or bf16 store ----
    const int doRelu = flags & FLAG_RELU;
    const int outBF  = flags & FLAG_BF16;
    #pragma unroll
    for (int i = 0; i < 2; ++i) {
        #pragma unroll
        for (int j = 0; j < 4; ++j) {
            const int col = colBase + wn * 64 + j * 16 + frow;
            if (col < Ncols) {
                const float bv = bias[col];
                #pragma unroll
                for (int v = 0; v < 8; ++v) {
                    const int row = rowBase + wm * 32 + i * 16 + v + 8 * half;
                    float r = acc[i][j][v] + bv;
                    if (doRelu) r = fmaxf(r, 0.f);
                    if (outBF) ((unsigned short*)Cv)[(size_t)row * ldC + colOff + col] = f2bf(r);
                    else       ((float*)Cv)[(size_t)row * ldC + colOff + col] = r;
                }
            }
        }
    }
}

// ---------------- helper kernels ----------------
__global__ void copy_pad_kernel(float* __restrict__ dst, const float* __restrict__ src,
                                int cols, int ld, int validRows) {
    const int r = blockIdx.y;
    const int c = blockIdx.x * blockDim.x + threadIdx.x;
    if (c < ld) dst[(size_t)r * ld + c] =
        (r < validRows && c < cols) ? src[(size_t)r * cols + c] : 0.f;
}

// f32 [rows x ld] -> bf16 [rows x ld]; rows >= validRows zeroed (keeps A pads NaN-free)
__global__ void cast_act_kernel(unsigned short* __restrict__ dst, const float* __restrict__ src,
                                int ld, int validRows) {
    const int r = blockIdx.y;
    const int c = blockIdx.x * blockDim.x + threadIdx.x;
    if (c < ld) dst[(size_t)r * ld + c] =
        (r < validRows) ? f2bf(src[(size_t)r * ld + c]) : (unsigned short)0;
}

// W[K x Ncols] f32 row-major -> Wt[colPad x ldW] bf16 transposed, zero-padded
__global__ void cast_wt_kernel(unsigned short* __restrict__ dst, const float* __restrict__ src,
                               int K, int Ncols, int ldW) {
    const int n = blockIdx.y;
    const int k = blockIdx.x * blockDim.x + threadIdx.x;
    if (k < ldW) dst[(size_t)n * ldW + k] =
        (n < Ncols && k < K) ? f2bf(src[(size_t)k * Ncols + n]) : (unsigned short)0;
}

__global__ void scale_kernel(float* __restrict__ hp, const float* __restrict__ h,
                             int ld, const float* __restrict__ epsp) {
    const float s = 1.0f + epsp[0];
    const int r = blockIdx.y;
    const int c = blockIdx.x * blockDim.x + threadIdx.x;
    if (c < ld) hp[(size_t)r * ld + c] = s * h[(size_t)r * ld + c];
}

__global__ void scatter_add_kernel(float* __restrict__ hp, const float* __restrict__ h,
                                   const int* __restrict__ ei, int E, int ld, int cols) {
    const int e = blockIdx.x;
    const int src = ei[e];
    const int dst = ei[E + e];
    const float* s = h + (size_t)src * ld;
    float* d = hp + (size_t)dst * ld;
    for (int c = threadIdx.x; c < cols; c += blockDim.x)
        atomicAdd(&d[c], s[c]);
}

__global__ void zero_kernel(float* __restrict__ p, int n) {
    const int i = blockIdx.x * blockDim.x + threadIdx.x;
    if (i < n) p[i] = 0.f;
}

__global__ void bn_stats_kernel(const float* __restrict__ H, int M, int ld, int cols,
                                float* __restrict__ sum, float* __restrict__ sq, int rowsPer) {
    const int c = blockIdx.y * blockDim.x + threadIdx.x;
    if (c >= cols) return;
    const int r0 = blockIdx.x * rowsPer;
    int r1 = r0 + rowsPer; if (r1 > M) r1 = M;
    float s = 0.f, q = 0.f;
    for (int r = r0; r < r1; ++r) {
        const float v = H[(size_t)r * ld + c];
        s += v; q += v * v;
    }
    atomicAdd(&sum[c], s);
    atomicAdd(&sq[c], q);
}

// in-place f32 BN (+ optional relu), valid rows only
__global__ void bn_apply_kernel(float* __restrict__ H, int ld, int cols,
                                const float* __restrict__ sum, const float* __restrict__ sq,
                                const float* __restrict__ g, const float* __restrict__ b,
                                float invN, int relu) {
    const int r = blockIdx.y;
    const int c = blockIdx.x * blockDim.x + threadIdx.x;
    if (c >= cols) return;
    const float mean = sum[c] * invN;
    const float var  = sq[c] * invN - mean * mean;
    float v = H[(size_t)r * ld + c];
    v = g[c] * (v - mean) * rsqrtf(var + BN_EPS) + b[c];
    if (relu) v = fmaxf(v, 0.f);
    H[(size_t)r * ld + c] = v;
}

// BN (+relu) reading f32, writing bf16 GEMM operand; pads (r>=valid or c>=cols) -> 0
__global__ void bn_apply_bf16_kernel(unsigned short* __restrict__ D, const float* __restrict__ H,
                                     int ld, int cols,
                                     const float* __restrict__ sum, const float* __restrict__ sq,
                                     const float* __restrict__ g, const float* __restrict__ b,
                                     float invN, int relu, int validRows) {
    const int r = blockIdx.y;
    const int c = blockIdx.x * blockDim.x + threadIdx.x;
    if (c >= ld) return;
    unsigned short o = 0;
    if (r < validRows && c < cols) {
        const float mean = sum[c] * invN;
        const float var  = sq[c] * invN - mean * mean;
        float v = H[(size_t)r * ld + c];
        v = g[c] * (v - mean) * rsqrtf(var + BN_EPS) + b[c];
        if (relu) v = fmaxf(v, 0.f);
        o = f2bf(v);
    }
    D[(size_t)r * ld + c] = o;
}

__global__ void pool_acc_kernel(float* __restrict__ pool, float* __restrict__ cnt,
                                const float* __restrict__ h, const int* __restrict__ batch,
                                int ld, int cols) {
    const int r = blockIdx.x;
    const int gi = batch[r];
    const float* s = h + (size_t)r * ld;
    float* d = pool + (size_t)gi * ld;
    for (int c = threadIdx.x; c < cols; c += blockDim.x)
        atomicAdd(&d[c], s[c]);
    if (threadIdx.x == 0) atomicAdd(&cnt[gi], 1.f);
}

// pooled mean -> bf16 GEMM operand (pads zeroed)
__global__ void pool_div_bf16_kernel(unsigned short* __restrict__ g, const float* __restrict__ pool,
                                     const float* __restrict__ cnt, int ld, int cols) {
    const int gi = blockIdx.y;
    const int c = blockIdx.x * blockDim.x + threadIdx.x;
    if (c < ld) {
        unsigned short o = 0;
        if (c < cols) {
            const float cn = fmaxf(cnt[gi], 1.f);
            o = f2bf(pool[(size_t)gi * ld + c] / cn);
        }
        g[(size_t)gi * ld + c] = o;
    }
}

// ---------------- launch ----------------
static inline int cdiv_i(int a, int b) { return (a + b - 1) / b; }

extern "C" void kernel_launch(void* const* d_in, const int* in_sizes, int n_in,
                              void* d_out, int out_size, void* d_ws, size_t ws_size,
                              hipStream_t stream) {
    const float* x        = (const float*)d_in[0];
    const int*   ei1      = (const int*)d_in[1];
    const int*   ei2      = (const int*)d_in[2];
    const int*   batch    = (const int*)d_in[3];
    const float* conv_eps = (const float*)d_in[4];
    const float* conv_W1  = (const float*)d_in[5];
    const float* conv_b1  = (const float*)d_in[6];
    const float* conv_bng = (const float*)d_in[7];
    const float* conv_bnb = (const float*)d_in[8];
    const float* conv_W2  = (const float*)d_in[9];
    const float* conv_b2  = (const float*)d_in[10];
    const float* mlp_W1   = (const float*)d_in[11];
    const float* mlp_b1   = (const float*)d_in[12];
    const float* mlp_W2   = (const float*)d_in[13];
    const float* mlp_b2   = (const float*)d_in[14];
    const float* bn_g     = (const float*)d_in[15];
    const float* bn_b     = (const float*)d_in[16];
    const float* head_W1  = (const float*)d_in[17];
    const float* head_b1  = (const float*)d_in[18];
    const float* head_W2  = (const float*)d_in[19];
    const float* head_b2  = (const float*)d_in[20];
    float* out = (float*)d_out;

    // ---- workspace carve-up (all 16B aligned) ----
    float* hbuf  = (float*)d_ws;                       // N_PAD x 704 (f32 node state)
    float* hp    = hbuf  + (size_t)N_PAD * LD7;        // N_PAD x 704
    float* wide  = hp    + (size_t)N_PAD * LD7;        // N_PAD x 1408 (pre-BN f32)
    float* stats = wide  + (size_t)N_PAD * LD14;       // 2 x 1408
    float* pool  = stats + 2 * LD14;                   // 512 x 704
    float* cnt   = pool  + (size_t)NGRAPH * LD7;       // 512
    unsigned short* hpbf   = (unsigned short*)(cnt + NGRAPH);
    unsigned short* widebf = hpbf   + (size_t)N_PAD * LD7;    // N_PAD x 1408
    unsigned short* hcatbf = widebf + (size_t)N_PAD * LD14;   // N_PAD x 1408
    unsigned short* tbf    = hcatbf + (size_t)N_PAD * LD14;   // N_PAD x 704
    unsigned short* gpoolbf= tbf    + (size_t)N_PAD * LD7;    // 512 x 704
    unsigned short* gheadbf= gpoolbf+ (size_t)NGRAPH * LD7;   // 512 x 704
    unsigned short* wtbuf  = gheadbf+ (size_t)NGRAPH * LD7;   // up to 1408 x 1408 bf16

    float* ssum = stats;
    float* ssq  = stats + LD14;
    const float invN = 1.0f / (float)N_NODES;

    auto gemm = [&](const unsigned short* A, int M, int K, int ldA, int ldW,
                    void* C, int ldC, int coff, int Ncols, const float* bias, int flags) {
        dim3 g(cdiv_i(M, BM), cdiv_i(Ncols, BN));
        gemm_bf16_kernel<<<g, 256, 0, stream>>>(A, ldA, wtbuf, ldW, C, ldC, coff, Ncols,
                                                bias, flags, cdiv_i(K, BK));
    };
    auto castWt = [&](const float* src, int K, int Ncols, int ldW) {
        const int padRows = cdiv_i(Ncols, BN) * BN;   // cover all async B-tile rows
        cast_wt_kernel<<<dim3(cdiv_i(ldW, 256), padRows), 256, 0, stream>>>(wtbuf, src, K, Ncols, ldW);
    };
    auto bnorm_bf16 = [&](unsigned short* D, const float* H, int ld, int cols,
                          const float* g, const float* b, int relu) {
        zero_kernel<<<cdiv_i(2 * LD14, 256), 256, 0, stream>>>(stats, 2 * LD14);
        bn_stats_kernel<<<dim3(cdiv_i(N_NODES, 256), cdiv_i(cols, 256)), 256, 0, stream>>>(
            H, N_NODES, ld, cols, ssum, ssq, 256);
        bn_apply_bf16_kernel<<<dim3(cdiv_i(ld, 256), N_PAD), 256, 0, stream>>>(
            D, H, ld, cols, ssum, ssq, g, b, invN, relu, N_NODES);
    };

    // h = pad(x) (zero pad rows + cols)
    copy_pad_kernel<<<dim3(cdiv_i(LD7, 256), N_PAD), 256, 0, stream>>>(hbuf, x, DIM, LD7, N_NODES);

    for (int l = 0; l < 5; ++l) {
        for (int j = 0; j < 2; ++j) {
            const int ci = 2 * l + j;
            const int* ei = (j == 0) ? ei1 : ei2;
            // hp = (1+eps)*h ; hp[dst] += h[src] ; cast to bf16
            scale_kernel<<<dim3(cdiv_i(LD7, 256), N_NODES), 256, 0, stream>>>(hp, hbuf, LD7, conv_eps + ci);
            scatter_add_kernel<<<N_EDGES, 256, 0, stream>>>(hp, hbuf, ei, N_EDGES, LD7, DIM);
            cast_act_kernel<<<dim3(cdiv_i(LD7, 256), N_PAD), 256, 0, stream>>>(hpbf, hp, LD7, N_NODES);
            // wide = hp @ W1 + b1  (f32 out: BN stats need it)
            castWt(conv_W1 + (size_t)ci * DIM * DIM2, DIM, DIM2, LD7);
            gemm(hpbf, N_PAD, DIM, LD7, LD7, wide, LD14, 0, DIM2, conv_b1 + (size_t)ci * DIM2, 0);
            // widebf = relu(bn(wide))  (bf16 GEMM operand)
            bnorm_bf16(widebf, wide, LD14, DIM2,
                       conv_bng + (size_t)ci * DIM2, conv_bnb + (size_t)ci * DIM2, FLAG_RELU);
            // hcatbf[:, j*702 : +702] = widebf @ W2 + b2  (bf16 out)
            castWt(conv_W2 + (size_t)ci * DIM2 * DIM, DIM2, DIM, LD14);
            gemm(widebf, N_PAD, DIM2, LD14, LD14, hcatbf, LD14, j * DIM, DIM,
                 conv_b2 + (size_t)ci * DIM, FLAG_BF16);
        }
        // tbf = relu(hcat @ mlpW1 + b1)  (bf16 out)
        castWt(mlp_W1 + (size_t)l * DIM2 * DIM, DIM2, DIM, LD14);
        gemm(hcatbf, N_PAD, DIM2, LD14, LD14, tbf, LD7, 0, DIM,
             mlp_b1 + (size_t)l * DIM, FLAG_RELU | FLAG_BF16);
        // h = tbf @ mlpW2 + b2 (f32: BN stats + scatter/pool need it)
        castWt(mlp_W2 + (size_t)l * DIM * DIM, DIM, DIM, LD7);
        gemm(tbf, N_PAD, DIM, LD7, LD7, hbuf, LD7, 0, DIM, mlp_b2 + (size_t)l * DIM, 0);
        // h = bn(h) (+ relu except last layer); bn_1 reused every layer (in-place f32)
        zero_kernel<<<cdiv_i(2 * LD14, 256), 256, 0, stream>>>(stats, 2 * LD14);
        bn_stats_kernel<<<dim3(cdiv_i(N_NODES, 256), cdiv_i(DIM, 256)), 256, 0, stream>>>(
            hbuf, N_NODES, LD7, DIM, ssum, ssq, 256);
        bn_apply_kernel<<<dim3(cdiv_i(DIM, 256), N_NODES), 256, 0, stream>>>(
            hbuf, LD7, DIM, ssum, ssq, bn_g, bn_b, invN, (l < 4) ? 1 : 0);
    }

    // global mean pool -> bf16 operand
    zero_kernel<<<cdiv_i(NGRAPH * LD7 + NGRAPH, 256), 256, 0, stream>>>(pool, NGRAPH * LD7 + NGRAPH);
    pool_acc_kernel<<<N_NODES, 256, 0, stream>>>(pool, cnt, hbuf, batch, LD7, DIM);
    pool_div_bf16_kernel<<<dim3(cdiv_i(LD7, 256), NGRAPH), 256, 0, stream>>>(gpoolbf, pool, cnt, LD7, DIM);

    // head
    castWt(head_W1, DIM, DIM, LD7);
    gemm(gpoolbf, NGRAPH, DIM, LD7, LD7, gheadbf, LD7, 0, DIM, head_b1, FLAG_RELU | FLAG_BF16);
    castWt(head_W2, DIM, NTASKS, LD7);
    gemm(gheadbf, NGRAPH, DIM, LD7, LD7, out, NTASKS, 0, NTASKS, head_b2, 0);
}
